// Multihead_attention_88424786690767
// MI455X (gfx1250) — matrix operand order repaired and checked by reading, not compiled, not run on hardware
//
#include <hip/hip_runtime.h>
#include <hip/hip_bf16.h>

typedef __attribute__((ext_vector_type(16))) __bf16 v16bf;
typedef __attribute__((ext_vector_type(8)))  __bf16 v8bf;
typedef __attribute__((ext_vector_type(8)))  float  v8f;

#define D_MODEL 1024
#define T_SEQ   2048
#define NHEADS  16
#define HDIM    64
#define BATCH   2
#define MROWS   (BATCH * T_SEQ)   // 4096 total rows

// ---------------------------------------------------------------------------
// WMMA helpers (CDNA5 wave32, 16x16x32 bf16 -> f32)
// ---------------------------------------------------------------------------
__device__ __forceinline__ v8f wmma_bf16(v16bf a, v16bf b, v8f c) {
  // 8 args: (neg_a, A, neg_b, B, c_mod, C, reuse_a, reuse_b)
  return __builtin_amdgcn_wmma_f32_16x16x32_bf16(
      false, a, false, b, (short)0, c, false, false);
}

__device__ __forceinline__ v8f vzero8() {
  v8f z;
#pragma unroll
  for (int i = 0; i < 8; ++i) z[i] = 0.0f;
  return z;
}

// A-operand (16x32 bf16) lane fragment.  Layout: lane (m = lane%16,
// half = lane/16) holds K(e) = e + 16*half, i.e. 16 contiguous
// elements at columns [16*half .. 16*half+15].
// `row` points at the start of this lane's 32-wide K-chunk of its A row.
__device__ __forceinline__ v16bf load_a_frag(const __bf16* row, int half) {
  v8bf lo = *(const v8bf*)(row + half * 16);
  v8bf hi = *(const v8bf*)(row + 8 + half * 16);
  v16bf r;
#pragma unroll
  for (int i = 0; i < 8; ++i) { r[i] = lo[i]; r[i + 8] = hi[i]; }
  return r;
}

// ---------------------------------------------------------------------------
// Convert / transpose-convert kernels
// ---------------------------------------------------------------------------
__global__ void cvt_f32_to_bf16(const float* __restrict__ in,
                                __bf16* __restrict__ out, int n) {
  int i = blockIdx.x * 256 + threadIdx.x;
  if (i < n) out[i] = (__bf16)in[i];
}

// out[n*1024 + k] = (bf16) in[k*1024 + n]   (1024x1024 weight transpose)
__global__ void tcvt_w(const float* __restrict__ in, __bf16* __restrict__ out) {
  int i = blockIdx.x * 256 + threadIdx.x;   // over 1M outputs
  int nn = i >> 10;
  int k  = i & 1023;
  out[i] = (__bf16)in[k * D_MODEL + nn];
}

// ---------------------------------------------------------------------------
// Generic bf16 WMMA GEMM:  C[M x 1024] = A[M x 1024] @ W + bias
//   A:  bf16 row-major [M x 1024]
//   BT: bf16 [1024 x 1024], row n holds column n of W (pre-transposed)
//   out_mode 0: bf16 out, [B,H,T,Dh] layout (Q / K)
//   out_mode 1: bf16 out, [B,H,Dh,T] layout (V transposed)
//   out_mode 2: f32  out, plain [M x 1024] row-major (final projection)
// Block = 256 threads = 8 waves; wave tile 16x32; block tile 64x64.
// grid.x = 1024/64 = 16 (N), grid.y = M/64 (rows)
// ---------------------------------------------------------------------------
__global__ void __launch_bounds__(256)
gemm_bf16_kernel(const __bf16* __restrict__ A, const __bf16* __restrict__ BT,
                 const float* __restrict__ bias, void* __restrict__ out,
                 int out_mode) {
  const int lane = threadIdx.x & 31;
  const int wave = threadIdx.x >> 5;
  const int ln16 = lane & 15;
  const int half = lane >> 4;
  const int m0 = blockIdx.y * 64 + (wave >> 1) * 16;
  const int n0 = blockIdx.x * 64 + (wave & 1) * 32;

  v8f c[2];
  c[0] = vzero8();
  c[1] = vzero8();

  const __bf16* arow = A + (size_t)(m0 + ln16) * D_MODEL;

#pragma unroll 4
  for (int kk = 0; kk < D_MODEL; kk += 32) {
    __builtin_prefetch(arow + kk + 256, 0, 0);   // global_prefetch_b8
    v16bf a = load_a_frag(arow + kk, half);
#pragma unroll
    for (int t = 0; t < 2; ++t) {
      // B (32x16) lane fragment: element e <-> K = e + 16*half, col = ln16.
      // BT row (n0+16t+ln16) is column (n0+16t+ln16) of W: contiguous 32B.
      const __bf16* brow =
          BT + (size_t)(n0 + 16 * t + ln16) * D_MODEL + kk + half * 16;
      v16bf bm = *(const v16bf*)brow;
      c[t] = wmma_bf16(a, bm, c[t]);
    }
  }

#pragma unroll
  for (int t = 0; t < 2; ++t) {
    const int col = n0 + 16 * t + ln16;
    const float bv = bias[col];
#pragma unroll
    for (int r = 0; r < 8; ++r) {
      const int mrow = m0 + r + 8 * half;   // C layout: row = r + 8*half
      const float val = c[t][r] + bv;
      if (out_mode == 2) {
        ((float*)out)[(size_t)mrow * D_MODEL + col] = val;
      } else {
        const int bb = mrow >> 11, trow = mrow & (T_SEQ - 1);
        const int hh = col >> 6, dh = col & (HDIM - 1);
        size_t idx;
        if (out_mode == 0)
          idx = ((size_t)(bb * NHEADS + hh) * T_SEQ + trow) * HDIM + dh;
        else
          idx = ((size_t)(bb * NHEADS + hh) * HDIM + dh) * T_SEQ + trow;
        ((__bf16*)out)[idx] = (__bf16)val;
      }
    }
  }
}

// ---------------------------------------------------------------------------
// Flash attention (causal) over bf16 Q/K/Vt with f32 online softmax.
//   Q, K : [B, H, T, Dh] bf16
//   Vt   : [B, H, Dh, T] bf16
//   ctx  : [B*T, 1024]   bf16 (head-merged, ready for output projection)
// Block = 256 threads = 8 independent waves; wave owns a 16-row q tile.
// grid.x = B*H*(T/128) = 512.  No cross-wave sync needed.
// ---------------------------------------------------------------------------
__global__ void __launch_bounds__(256)
attn_kernel(const __bf16* __restrict__ Q, const __bf16* __restrict__ K,
            const __bf16* __restrict__ Vt, __bf16* __restrict__ ctx) {
  __shared__ __align__(32) __bf16 plds[8 * 16 * 32];   // 1 KB per wave

  const int lane = threadIdx.x & 31;
  const int wave = threadIdx.x >> 5;
  const int ln16 = lane & 15;
  const int half = lane >> 4;

  const int qblk = blockIdx.x & 15;        // T/128 = 16 q-blocks
  const int bh   = blockIdx.x >> 4;        // 0..31 (b*16+h)
  const int b    = bh >> 4;
  const int h    = bh & 15;
  const int qb   = qblk * 128 + wave * 16; // this wave's q-tile base row

  // Q A-operand fragments for Dh = 64 -> two 32-wide K-chunks.
  const __bf16* qrow = Q + ((size_t)bh * T_SEQ + qb + ln16) * HDIM;
  v16bf aQ0 = load_a_frag(qrow + 0,  half);
  v16bf aQ1 = load_a_frag(qrow + 32, half);

  float mrow[8], lrow[8];
  v8f o[4];
#pragma unroll
  for (int r = 0; r < 8; ++r) { mrow[r] = -__builtin_inff(); lrow[r] = 0.0f; }
#pragma unroll
  for (int j = 0; j < 4; ++j) o[j] = vzero8();

  __bf16* myp = plds + wave * (16 * 32);
  const float sc = 0.125f;                 // 1/sqrt(Dh)

  for (int k0 = 0; k0 <= qb + 15; k0 += 32) {
    // ---- scores S[16 x 32] = Q_tile @ K[k0:k0+32]^T (two 16-col tiles) ----
    v8f s[2];
#pragma unroll
    for (int t = 0; t < 2; ++t) {
      s[t] = vzero8();
      // B operand of scores: B[d][k'] = K[k'][d]; lane reads a contiguous
      // 16-element slice of key row (k0+16t+ln16) at d-offset half*16 (+32).
      const __bf16* krow =
          K + ((size_t)bh * T_SEQ + k0 + 16 * t + ln16) * HDIM + half * 16;
      v16bf bk0 = *(const v16bf*)(krow);
      v16bf bk1 = *(const v16bf*)(krow + 32);
      s[t] = wmma_bf16(aQ0, bk0, s[t]);
      s[t] = wmma_bf16(aQ1, bk1, s[t]);
    }

    // ---- causal mask + scale + online softmax (row = r + 8*half) ----
#pragma unroll
    for (int r = 0; r < 8; ++r) {
      const int qr = qb + r + 8 * half;
      float v0 = (k0 + ln16      <= qr) ? s[0][r] * sc : -__builtin_inff();
      float v1 = (k0 + 16 + ln16 <= qr) ? s[1][r] * sc : -__builtin_inff();
      float vm = fmaxf(v0, v1);
#pragma unroll
      for (int off = 1; off < 16; off <<= 1)
        vm = fmaxf(vm, __shfl_xor(vm, off, 32));   // stays within half-group
      const float mnew  = fmaxf(mrow[r], vm);
      const float alpha = __expf(mrow[r] - mnew);  // first step: exp(-inf)=0
      const float p0 = __expf(v0 - mnew);
      const float p1 = __expf(v1 - mnew);
      float rs = p0 + p1;
#pragma unroll
      for (int off = 1; off < 16; off <<= 1)
        rs += __shfl_xor(rs, off, 32);
      lrow[r] = lrow[r] * alpha + rs;
      mrow[r] = mnew;
#pragma unroll
      for (int j = 0; j < 4; ++j) o[j][r] *= alpha;
      // stage P into this wave's LDS slice (C layout -> row-major 16x32)
      myp[(r + 8 * half) * 32 + ln16]      = (__bf16)p0;
      myp[(r + 8 * half) * 32 + 16 + ln16] = (__bf16)p1;
    }

    asm volatile("s_wait_dscnt 0" ::: "memory");   // same-wave LDS RAW

    // ---- O += P @ V : reload P in A layout, V B-operand from Vt rows ----
    v16bf pa = load_a_frag(myp + ln16 * 32, half);
#pragma unroll
    for (int j = 0; j < 4; ++j) {
      const __bf16* vrow =
          Vt + ((size_t)bh * HDIM + 16 * j + ln16) * T_SEQ + k0 + half * 16;
      v16bf bv = *(const v16bf*)vrow;
      o[j] = wmma_bf16(pa, bv, o[j]);
    }
  }

  // ---- normalize and write ctx (head-merged [B*T, 1024]) ----
#pragma unroll
  for (int r = 0; r < 8; ++r) {
    const float inv = 1.0f / lrow[r];
    const int trow = qb + r + 8 * half;
    const size_t orow = ((size_t)b * T_SEQ + trow) * D_MODEL + h * HDIM;
#pragma unroll
    for (int j = 0; j < 4; ++j)
      ctx[orow + 16 * j + ln16] = (__bf16)(o[j][r] * inv);
  }
}

// ---------------------------------------------------------------------------
// Host-side launch
// ---------------------------------------------------------------------------
extern "C" void kernel_launch(void* const* d_in, const int* in_sizes, int n_in,
                              void* d_out, int out_size, void* d_ws,
                              size_t ws_size, hipStream_t stream) {
  const float* x  = (const float*)d_in[0];
  const float* Wq = (const float*)d_in[1];
  const float* bq = (const float*)d_in[2];
  const float* Wk = (const float*)d_in[3];
  const float* bk = (const float*)d_in[4];
  const float* Wv = (const float*)d_in[5];
  const float* bv = (const float*)d_in[6];
  const float* Wo = (const float*)d_in[7];
  const float* bo = (const float*)d_in[8];

  char* ws = (char*)d_ws;
  size_t off = 0;
  __bf16* xb   = (__bf16*)(ws + off); off += (size_t)MROWS * D_MODEL * 2;   // 8 MB
  __bf16* WqT  = (__bf16*)(ws + off); off += (size_t)D_MODEL * D_MODEL * 2; // 2 MB
  __bf16* WkT  = (__bf16*)(ws + off); off += (size_t)D_MODEL * D_MODEL * 2;
  __bf16* WvT  = (__bf16*)(ws + off); off += (size_t)D_MODEL * D_MODEL * 2;
  __bf16* WoT  = (__bf16*)(ws + off); off += (size_t)D_MODEL * D_MODEL * 2;
  __bf16* Qt   = (__bf16*)(ws + off); off += (size_t)MROWS * D_MODEL * 2;   // 8 MB
  __bf16* Kt   = (__bf16*)(ws + off); off += (size_t)MROWS * D_MODEL * 2;
  __bf16* Vt   = (__bf16*)(ws + off); off += (size_t)MROWS * D_MODEL * 2;
  __bf16* ctxb = (__bf16*)(ws + off); off += (size_t)MROWS * D_MODEL * 2;

  // 1) precision conversion / weight transposition
  const int nx = MROWS * D_MODEL;
  cvt_f32_to_bf16<<<nx / 256, 256, 0, stream>>>(x, xb, nx);
  const int nw = D_MODEL * D_MODEL;
  tcvt_w<<<nw / 256, 256, 0, stream>>>(Wq, WqT);
  tcvt_w<<<nw / 256, 256, 0, stream>>>(Wk, WkT);
  tcvt_w<<<nw / 256, 256, 0, stream>>>(Wv, WvT);
  tcvt_w<<<nw / 256, 256, 0, stream>>>(Wo, WoT);

  // 2) QKV projections (WMMA)
  dim3 gg(D_MODEL / 64, MROWS / 64);   // 16 x 64
  dim3 bb(256);
  gemm_bf16_kernel<<<gg, bb, 0, stream>>>(xb, WqT, bq, (void*)Qt, 0);
  gemm_bf16_kernel<<<gg, bb, 0, stream>>>(xb, WkT, bk, (void*)Kt, 0);
  gemm_bf16_kernel<<<gg, bb, 0, stream>>>(xb, WvT, bv, (void*)Vt, 1);

  // 3) causal flash attention (WMMA)
  attn_kernel<<<BATCH * NHEADS * (T_SEQ / 128), 256, 0, stream>>>(Qt, Kt, Vt,
                                                                  ctxb);

  // 4) output projection -> f32 d_out (WMMA)
  gemm_bf16_kernel<<<gg, bb, 0, stream>>>(ctxb, WoT, bo, d_out, 2);
}
